// LSTMmodel_62354335204006
// MI455X (gfx1250) — compile-verified, matrix operand order checked by reading, not run on hardware
//
#include <hip/hip_runtime.h>
#include <hip/hip_bf16.h>

// ---------------------------------------------------------------------------
// 2-layer window-LSTM rollout for MI455X (gfx1250), wave32 + WMMA bf16.
//   B=128 batch, H=128 hidden, WS=4 window, T=256, OUT=2, 252 predictions.
// Single persistent 1024-thread workgroup (32 waves on one WGP):
//   - h0/h1 ping-pong in LDS as bf16, c kept in registers (WMMA D-layout).
//   - layer-0 input projection + bias done by WMMA via K=32-padded input.
//   - weight fragments streamed from L2 each k-step; a laundered 0-offset
//     (not a laundered pointer!) blocks LICM hoist/spill while preserving
//     global address-space provenance -> global_load_b128, not flat_load.
//   - gate nonlinearities on the CDNA5 V_TANH_F32 trans unit.
// ---------------------------------------------------------------------------

typedef __bf16 bf16_t;
typedef __attribute__((ext_vector_type(16))) __bf16 v16bf;
typedef __attribute__((ext_vector_type(8)))  __bf16 v8bf;
typedef __attribute__((ext_vector_type(8)))  float  v8f;

#define BATCH 128
#define HID   128
#define G4    512      // 4*HID gate width
#define WSZ   4
#define TLEN  256
#define NOUT  252      // T - WS predictions
#define XK    32       // padded layer-0 K (3 inputs + 1.0 bias channel + zeros)
#define OUT_OFF_H 64512            // 128*252*2
#define OUT_OFF_C 97280            // + 2*128*128

#if __has_builtin(__builtin_amdgcn_tanhf)
__device__ __forceinline__ float fast_tanh(float x) { return __builtin_amdgcn_tanhf(x); }
#else
__device__ __forceinline__ float fast_tanh(float x) { return tanhf(x); }
#endif

__device__ __forceinline__ float fast_sigmoid(float x) {
    return 0.5f * fast_tanh(0.5f * x) + 0.5f;   // sigmoid via V_TANH_F32
}

// Opaque zero offset (uniform, in an SGPR): loads addressed through
// (base + launder0()) cannot be CSE'd/hoisted across k-steps, but the base
// pointer keeps its global-memory provenance, so loads stay global_load_b128.
__device__ __forceinline__ int launder0() {
    int v = 0;
    asm volatile("" : "+s"(v));
    return v;
}

__device__ __forceinline__ v8f wmma_bf16(v16bf a, v16bf b, v8f c) {
    // D = A(16x32 bf16) * B(32x16 bf16) + C(16x16 f32)
    return __builtin_amdgcn_wmma_f32_16x16x32_bf16(
        /*neg_a=*/false, a, /*neg_b=*/false, b,
        /*c_mod=*/(short)0, c, /*reuse_a=*/false, /*reuse_b=*/false);
}

// A fragment (16x32 bf16) from row-major bf16 [128][stride] in LDS.
// ISA layout: lanes 0-15 row M=lane%16, K in {0..7, 16..23}; lanes 16-31 same
// M, K in {8..15, 24..31}. Two contiguous 16B LDS loads per lane.
__device__ __forceinline__ v16bf loadA_s(const bf16_t* src, int stride, int mt,
                                         int ks, int lane) {
    const int m   = mt * 16 + (lane & 15);
    const int off = (lane < 16) ? 0 : 8;
    const bf16_t* base = src + m * stride + ks * 32;
    union { v16bf v; v8bf h[2]; } u;
    u.h[0] = *(const v8bf*)(base + off);
    u.h[1] = *(const v8bf*)(base + 16 + off);
    return u.v;
}

// B fragment: prep kernel packed lane-contiguous 16 halves (32B) per lane.
__device__ __forceinline__ v16bf loadB(const bf16_t* frags, int fid, int lane) {
    return *(const v16bf*)(frags + ((fid * 32) + lane) * 16);
}

// half-index -> K mapping inside one 32-wide WMMA K-step (B matrix, per lane)
__device__ __forceinline__ int kmap(int lane, int h) {
    return (lane < 16) ? ((h < 8) ? h : h + 8)
                       : ((h < 8) ? h + 8 : h + 16);
}

// ---------------------------------------------------------------------------
// Prep: pack Whh0 / Wih1 / Whh1 ([512,128] f32 row-major) into bf16 B-matrix
// fragments, pack [Wih0 | b0] into K=32 fragments, fold layer-1 bias.
// ---------------------------------------------------------------------------
__global__ void lstm_prep_kernel(const float* __restrict__ Wih0,
                                 const float* __restrict__ Whh0,
                                 const float* __restrict__ Wih1,
                                 const float* __restrict__ Whh1,
                                 const float* __restrict__ bih0,
                                 const float* __restrict__ bhh0,
                                 const float* __restrict__ bih1,
                                 const float* __restrict__ bhh1,
                                 bf16_t* __restrict__ fWhh0,
                                 bf16_t* __restrict__ fWih1,
                                 bf16_t* __restrict__ fWhh1,
                                 bf16_t* __restrict__ fWih0p,
                                 float* __restrict__ b1) {
    const int idx = blockIdx.x * 256 + threadIdx.x;
    if (idx < 3 * 65536) {
        const int mat  = idx >> 16;
        const int r    = idx & 65535;
        const int f    = r >> 9;          // fragment id: ntile*4 + ks
        const int lane = (r >> 4) & 31;
        const int h    = r & 15;
        const int n = (f >> 2) * 16 + (lane & 15);
        const int k = (f & 3) * 32 + kmap(lane, h);
        const float* W  = (mat == 0) ? Whh0 : (mat == 1) ? Wih1 : Whh1;
        bf16_t* dst     = (mat == 0) ? fWhh0 : (mat == 1) ? fWih1 : fWhh1;
        dst[r] = (bf16_t)W[n * HID + k];
    } else if (idx < 3 * 65536 + 16384) {
        // [Wih0 | b0] padded to K=32: k<3 -> Wih0[n][k], k==3 -> b0[n], else 0
        const int r    = idx - 3 * 65536;
        const int f    = r >> 9;          // ntile 0..31 (single K-step)
        const int lane = (r >> 4) & 31;
        const int h    = r & 15;
        const int n = f * 16 + (lane & 15);
        const int k = kmap(lane, h);
        float v = 0.0f;
        if (k < 3)       v = Wih0[n * 3 + k];
        else if (k == 3) v = bih0[n] + bhh0[n];
        fWih0p[r] = (bf16_t)v;
    } else if (idx < 3 * 65536 + 16384 + G4) {
        const int j = idx - 3 * 65536 - 16384;
        b1[j] = bih1[j] + bhh1[j];
    }
}

// ---------------------------------------------------------------------------
// Persistent rollout kernel: one workgroup, 1024 threads (32 waves).
// ---------------------------------------------------------------------------
__global__ __launch_bounds__(1024, 1)
void lstm_rollout_kernel(const float* __restrict__ traj,    // [128,256,3]
                         const float* __restrict__ Wl,      // [2,128]
                         const float* __restrict__ bl,      // [2]
                         const bf16_t* __restrict__ fWhh0,
                         const bf16_t* __restrict__ fWih1,
                         const bf16_t* __restrict__ fWhh1,
                         const bf16_t* __restrict__ fWih0p,
                         const float* __restrict__ b1,
                         float* __restrict__ out) {
    extern __shared__ char smem[];
    bf16_t* h0A  = (bf16_t*)smem;                 // [128][128] bf16 ping
    bf16_t* h0B  = h0A + BATCH * HID;             //            pong
    bf16_t* h1A  = h0B + BATCH * HID;
    bf16_t* h1B  = h1A + BATCH * HID;
    bf16_t* xpad = h1B + BATCH * HID;             // [4][128][32] padded input
    float*  ring = (float*)(xpad + WSZ * BATCH * XK);  // [4][128][2]

    const int tid   = threadIdx.x;
    const int lane  = tid & 31;
    const int wave  = tid >> 5;
    const int nloc  = lane & 15;        // D-tile column within 16
    const int rhi   = (lane >> 4) * 8;  // D-tile row offset (+8 for hi lanes)

    float c0r[2][8];                    // layer-0 cell state (per pair)
    float c1r[2][8];                    // layer-1 cell state

    // constant part of padded input: channel 3 = 1.0 (bias), 4..31 = 0
    for (int i = tid; i < WSZ * BATCH * XK; i += 1024) {
        const int k = i & (XK - 1);
        xpad[i] = (bf16_t)((k == 3) ? 1.0f : 0.0f);
    }
    __syncthreads();

    for (int s = 0; s < NOUT; ++s) {
        // ---- build window input channels 0..2 (phase-dependent) ----
        if (tid < WSZ * BATCH) {
            const int t = tid >> 7, b = tid & 127;
            float v0, v1, v2;
            if (s == 0) {
                const float* tp = &traj[(b * TLEN + t) * 3];
                v0 = tp[0]; v1 = tp[1]; v2 = tp[2];
            } else if (s < WSZ) {
                if (t < 3) {
                    const float* tp = &traj[(b * TLEN + (s + t)) * 3];
                    v0 = tp[0]; v1 = tp[1]; v2 = tp[2];
                } else {  // predicted row: [ctrl, out0, out1]
                    v0 = traj[(b * TLEN + (s + 4)) * 3 + 0];
                    const float* rp = &ring[((s - 1) & 3) * 256 + b * 2];
                    v1 = rp[0]; v2 = rp[1];
                }
            } else {      // autoregressive: [out0, out1, ctrl]
                const float* rp = &ring[((s - 4 + t) & 3) * 256 + b * 2];
                v0 = rp[0]; v1 = rp[1];
                v2 = traj[(b * TLEN + (s + t)) * 3 + 0];
            }
            bf16_t* xp = &xpad[(t * BATCH + b) * XK];
            xp[0] = (bf16_t)v0; xp[1] = (bf16_t)v1; xp[2] = (bf16_t)v2;
        }
        // ---- zero hidden state (window-reset LSTM) ----
        for (int i = tid; i < BATCH * HID; i += 1024) {
            h0A[i] = (bf16_t)0.0f;
            h1A[i] = (bf16_t)0.0f;
        }
        #pragma unroll
        for (int p = 0; p < 2; ++p)
            #pragma unroll
            for (int r = 0; r < 8; ++r) { c0r[p][r] = 0.0f; c1r[p][r] = 0.0f; }
        __syncthreads();

        for (int t = 0; t < WSZ; ++t) {
            const bf16_t* s0 = (t & 1) ? h0B : h0A;   // layer-0 h (read)
            bf16_t*       d0 = (t & 1) ? h0A : h0B;   // layer-0 h (write)
            const bf16_t* s1 = (t & 1) ? h1B : h1A;   // layer-1 h (read)
            bf16_t*       d1 = (t & 1) ? h1A : h1B;   // layer-1 h (write)
            const bf16_t* xt = xpad + t * BATCH * XK;

            // ================= layer 0 (pure WMMA: input proj + recurrent) ==
            #pragma unroll
            for (int p = 0; p < 2; ++p) {
                const int pair = wave * 2 + p;
                const int mt = pair >> 3, jt = pair & 7;
                v8f acc[4];
                {   // xg0 = [x,1] @ [Wih0|b0]^T   (one K=32 WMMA per gate)
                    const bf16_t* wb = fWih0p + launder0();
                    const v16bf a = loadA_s(xt, XK, mt, 0, lane);
                    #pragma unroll
                    for (int g = 0; g < 4; ++g) {
                        v8f z;
                        #pragma unroll
                        for (int r = 0; r < 8; ++r) z[r] = 0.0f;
                        acc[g] = wmma_bf16(a, loadB(wb, g * 8 + jt, lane), z);
                    }
                }
                #pragma unroll
                for (int ks = 0; ks < 4; ++ks) {   // += h0 @ Whh0^T
                    const bf16_t* wb = fWhh0 + launder0();
                    const v16bf a = loadA_s(s0, HID, mt, ks, lane);
                    #pragma unroll
                    for (int g = 0; g < 4; ++g)
                        acc[g] = wmma_bf16(a, loadB(wb, (g * 8 + jt) * 4 + ks, lane), acc[g]);
                }
                // gates (PyTorch order i,f,g,o) — all in-lane, write new h
                #pragma unroll
                for (int r = 0; r < 8; ++r) {
                    const float ig = fast_sigmoid(acc[0][r]);
                    const float fg = fast_sigmoid(acc[1][r]);
                    const float gg = fast_tanh(acc[2][r]);
                    const float og = fast_sigmoid(acc[3][r]);
                    const float c  = fg * c0r[p][r] + ig * gg;
                    c0r[p][r] = c;
                    d0[(mt * 16 + rhi + r) * HID + jt * 16 + nloc] =
                        (bf16_t)(og * fast_tanh(c));
                }
            }
            __syncthreads();   // new h0 (= y0_t) visible to all waves

            // ================= layer 1: b1 + y0_t @ Wih1^T + h1 @ Whh1^T ====
            #pragma unroll
            for (int p = 0; p < 2; ++p) {
                const int pair = wave * 2 + p;
                const int mt = pair >> 3, jt = pair & 7;
                v8f acc[4];
                #pragma unroll
                for (int g = 0; g < 4; ++g)
                    #pragma unroll
                    for (int r = 0; r < 8; ++r) acc[g][r] = 0.0f;
                #pragma unroll
                for (int ks = 0; ks < 4; ++ks) {   // input: y0_t (new h0)
                    const bf16_t* wb = fWih1 + launder0();
                    const v16bf a = loadA_s(d0, HID, mt, ks, lane);
                    #pragma unroll
                    for (int g = 0; g < 4; ++g)
                        acc[g] = wmma_bf16(a, loadB(wb, (g * 8 + jt) * 4 + ks, lane), acc[g]);
                }
                #pragma unroll
                for (int ks = 0; ks < 4; ++ks) {   // recurrent: h1
                    const bf16_t* wb = fWhh1 + launder0();
                    const v16bf a = loadA_s(s1, HID, mt, ks, lane);
                    #pragma unroll
                    for (int g = 0; g < 4; ++g)
                        acc[g] = wmma_bf16(a, loadB(wb, (g * 8 + jt) * 4 + ks, lane), acc[g]);
                }
                const float bi = b1[0 * HID + jt * 16 + nloc];
                const float bf = b1[1 * HID + jt * 16 + nloc];
                const float bg = b1[2 * HID + jt * 16 + nloc];
                const float bo = b1[3 * HID + jt * 16 + nloc];
                #pragma unroll
                for (int r = 0; r < 8; ++r) {
                    const float ig = fast_sigmoid(acc[0][r] + bi);
                    const float fg = fast_sigmoid(acc[1][r] + bf);
                    const float gg = fast_tanh(acc[2][r] + bg);
                    const float og = fast_sigmoid(acc[3][r] + bo);
                    const float c  = fg * c1r[p][r] + ig * gg;
                    c1r[p][r] = c;
                    d1[(mt * 16 + rhi + r) * HID + jt * 16 + nloc] =
                        (bf16_t)(og * fast_tanh(c));
                }
            }
            __syncthreads();   // new h1 visible; also guards next t's h0 read
        }

        // ---- head: out[:, s] = base + h1_last @ Wl^T + bl ----
        // after t=3 the final h lives in the A (ping) buffers
        if (tid < 2 * BATCH) {
            const int b = tid >> 1, o = tid & 1;
            float acc = bl[o];
            const float* wl = &Wl[o * HID];
            for (int k = 0; k < HID; ++k)
                acc += (float)h1A[b * HID + k] * wl[k];
            const float base = (s < WSZ)
                ? traj[(b * TLEN + (s + 3)) * 3 + 1 + o]
                : ring[((s - 1) & 3) * 256 + b * 2 + o];
            const float val = base + acc;
            out[(b * NOUT + s) * 2 + o] = val;
            ring[(s & 3) * 256 + b * 2 + o] = val;
        }

        // ---- final hidden/cell state dump (matches torch (h_n, c_n)) ----
        if (s == NOUT - 1) {
            __syncthreads();
            for (int i = tid; i < BATCH * HID; i += 1024) {
                out[OUT_OFF_H + i]               = (float)h0A[i];
                out[OUT_OFF_H + BATCH * HID + i] = (float)h1A[i];
            }
            #pragma unroll
            for (int p = 0; p < 2; ++p) {
                const int pair = wave * 2 + p;
                const int mt = pair >> 3, jt = pair & 7;
                #pragma unroll
                for (int r = 0; r < 8; ++r) {
                    const int m = mt * 16 + rhi + r;
                    const int n = jt * 16 + nloc;
                    out[OUT_OFF_C + m * HID + n]               = c0r[p][r];
                    out[OUT_OFF_C + BATCH * HID + m * HID + n] = c1r[p][r];
                }
            }
        }
        __syncthreads();   // ring visible before next window build / h re-zero
    }
}

// ---------------------------------------------------------------------------
extern "C" void kernel_launch(void* const* d_in, const int* in_sizes, int n_in,
                              void* d_out, int out_size, void* d_ws, size_t ws_size,
                              hipStream_t stream) {
    (void)in_sizes; (void)n_in; (void)out_size; (void)ws_size;
    const float* traj = (const float*)d_in[0];
    const float* Wih0 = (const float*)d_in[1];
    const float* Whh0 = (const float*)d_in[2];
    const float* bih0 = (const float*)d_in[3];
    const float* bhh0 = (const float*)d_in[4];
    const float* Wih1 = (const float*)d_in[5];
    const float* Whh1 = (const float*)d_in[6];
    const float* bih1 = (const float*)d_in[7];
    const float* bhh1 = (const float*)d_in[8];
    const float* Wl   = (const float*)d_in[9];
    const float* bl   = (const float*)d_in[10];

    bf16_t* fWhh0  = (bf16_t*)d_ws;                  // 65536 halves each
    bf16_t* fWih1  = fWhh0 + 65536;
    bf16_t* fWhh1  = fWih1 + 65536;
    bf16_t* fWih0p = fWhh1 + 65536;                  // 16384 halves
    float*  b1     = (float*)(fWih0p + 16384);       // 512 floats

    const int prep_threads = 3 * 65536 + 16384 + G4;
    lstm_prep_kernel<<<(prep_threads + 255) / 256, 256, 0, stream>>>(
        Wih0, Whh0, Wih1, Whh1, bih0, bhh0, bih1, bhh1,
        fWhh0, fWih1, fWhh1, fWih0p, b1);

    const size_t lds = (size_t)(4 * BATCH * HID) * sizeof(bf16_t)   // h ping-pong
                     + (size_t)(WSZ * BATCH * XK) * sizeof(bf16_t)  // padded x
                     + (size_t)(4 * BATCH * 2) * sizeof(float);     // pred ring
    lstm_rollout_kernel<<<1, 1024, lds, stream>>>(
        traj, Wl, bl, fWhh0, fWih1, fWhh1, fWih0p, b1, (float*)d_out);
}
